// DaVinciMLP_7808250544921
// MI455X (gfx1250) — matrix operand order repaired and checked by reading, not compiled, not run on hardware
//
#include <hip/hip_runtime.h>

// ---------------------------------------------------------------- constants
#define N_TOK  8192
#define HID    5120
#define INTER  20480
#define NE     3

typedef __bf16 bf16_t;
typedef bf16_t v16bf __attribute__((ext_vector_type(16)));
typedef float  v8f   __attribute__((ext_vector_type(8)));

union FragB32 { v16bf v; uint4 q[2]; };

#define BM 128
#define BN 128
#define BK 32
#define THREADS 256   // 8 wave32s
#define NBUF 3        // async ring depth
#define BUFB 8192u    // bytes per A (or B) stage buffer

// CDNA5 async global->LDS copy (16B per lane), tracked by ASYNCcnt.
#define ASYNC_CP_B128(lds, gaddr)                                         \
    asm volatile("global_load_async_to_lds_b128 %0, %1, off"              \
                 :: "v"(lds), "v"(gaddr) : "memory")

// ASYNCcnt completes in order: waiting <=4 retires the oldest stage (4 ops)
// while the next stage's 4 ops stay in flight.
#define WAIT_ASYNC(n) asm volatile("s_wait_asynccnt " #n ::: "memory")

// ---------------------------------------------------------------- helpers
__global__ __launch_bounds__(32) void init_counts(int* counts) {
    if (threadIdx.x < NE) counts[threadIdx.x] = 0;
}

__global__ __launch_bounds__(256) void count_kernel(const int* __restrict__ mod,
                                                    int* __restrict__ counts) {
    int i = blockIdx.x * 256 + threadIdx.x;
    if (i < N_TOK) atomicAdd(&counts[mod[i]], 1);
}

__global__ __launch_bounds__(32) void scan_kernel(const int* __restrict__ counts,
                                                  int* __restrict__ offs,
                                                  int* __restrict__ cursor) {
    if (threadIdx.x == 0) {
        int a = counts[0], b = counts[1];
        offs[0] = 0; offs[1] = a; offs[2] = a + b;
        cursor[0] = 0; cursor[1] = a; cursor[2] = a + b;
    }
}

// Fused per-token ModalityRMSNorm + gather into expert-contiguous layout.
__global__ __launch_bounds__(256) void rmsnorm_gather(
    const bf16_t* __restrict__ x,
    const int*    __restrict__ mod,
    const bf16_t* __restrict__ norm_w,
    bf16_t*       __restrict__ xn,      // [N_TOK, HID] gathered
    int*          __restrict__ cursor,
    int*          __restrict__ perm)
{
    __shared__ float red[256];
    __shared__ int s_pos, s_e;
    const int t = blockIdx.x;             // token id
    if (threadIdx.x == 0) {
        int e   = mod[t];
        int pos = atomicAdd(&cursor[e], 1);
        perm[pos] = t;
        s_pos = pos; s_e = e;
    }
    __syncthreads();
    const int e = s_e, pos = s_pos;

    const bf16_t* xr = x + (size_t)t * HID;
    float ss = 0.0f;
    for (int i = threadIdx.x * 8; i < HID; i += 256 * 8) {
        union { uint4 q; bf16_t h[8]; } u;
        u.q = *(const uint4*)(xr + i);
#pragma unroll
        for (int c = 0; c < 8; ++c) { float f = (float)u.h[c]; ss += f * f; }
    }
    red[threadIdx.x] = ss;
    __syncthreads();
    for (int s = 128; s > 0; s >>= 1) {
        if (threadIdx.x < s) red[threadIdx.x] += red[threadIdx.x + s];
        __syncthreads();
    }
    const float scale = rsqrtf(red[0] * (1.0f / (float)HID) + 1e-6f);

    const bf16_t* wr = norm_w + (size_t)e * HID;
    bf16_t* orow = xn + (size_t)pos * HID;
    for (int i = threadIdx.x * 8; i < HID; i += 256 * 8) {
        union { uint4 q; bf16_t h[8]; } ux, uw, uo;
        ux.q = *(const uint4*)(xr + i);
        uw.q = *(const uint4*)(wr + i);
#pragma unroll
        for (int c = 0; c < 8; ++c) {
            float f = (float)ux.h[c];
            float w = (float)uw.h[c] + 1.0f;
            uo.h[c] = (bf16_t)(f * scale * w);
        }
        *(uint4*)(orow + i) = uo.q;
    }
}

// ---------------------------------------------------------------- WMMA GEMM
// C[m,n] = sum_k A[m,k] * Bw[e][n,k].  3-stage async ring in LDS.
// Steady state is branch-free: wait asynccnt<=4 -> barrier -> issue s+2 ->
// 12x ds_load_b128 -> 8x wmma.  Two drain stages are peeled after the loop.
template <bool GELU>
__global__ __launch_bounds__(THREADS) void moe_gemm(
    const bf16_t* __restrict__ A,      // [N_TOK, K] expert-gathered rows
    const bf16_t* __restrict__ Bw,     // [NE * browsPerE, K]
    bf16_t*       __restrict__ Out,
    const int*    __restrict__ offs,
    const int*    __restrict__ counts,
    const int*    __restrict__ perm,
    int K, int ONum, int browsPerE)
{
    __shared__ uint4 AsQ[NBUF][BM * BK / 8];   // 3 x 8 KB
    __shared__ uint4 BsQ[NBUF][BN * BK / 8];   // 3 x 8 KB

    const int e    = blockIdx.z;
    const int base = offs[e];
    const int cnt  = counts[e];
    const int m0   = blockIdx.y * BM;
    if (m0 >= cnt) return;               // block-uniform early exit
    const int n0   = blockIdx.x * BN;

    const int tid  = threadIdx.x;
    const int wave = tid >> 5;
    const int lane = tid & 31;
    const int wm   = wave >> 2;          // 0..1  (64-row slab)
    const int wn   = wave & 3;           // 0..3  (32-col slab)
    const int hl   = lane >> 4;          // lane half
    const int lr   = lane & 15;

    const bf16_t* Bbase = Bw + (size_t)e * (size_t)browsPerE * (size_t)K;

    const bf16_t* aptr[2];
    const bf16_t* bptr[2];
#pragma unroll
    for (int p = 0; p < 2; ++p) {
        long r = (long)base + m0 + (tid >> 2) + p * 64;
        if (r >= N_TOK) r = 0;           // clamp (garbage rows masked at store)
        aptr[p] = A + (size_t)r * K + (tid & 3) * 8;
        bptr[p] = Bbase + (size_t)(n0 + (tid >> 2) + p * 64) * K + (tid & 3) * 8;
    }

    // Generic __shared__ pointer low 32 bits == LDS byte offset (aperture form).
    const unsigned int ldsA = (unsigned int)(uintptr_t)&AsQ[0][tid];
    const unsigned int ldsB = (unsigned int)(uintptr_t)&BsQ[0][tid];

    v8f acc[4][2];
#pragma unroll
    for (int i = 0; i < 4; ++i)
#pragma unroll
        for (int j = 0; j < 2; ++j)
#pragma unroll
            for (int r = 0; r < 8; ++r) acc[i][j][r] = 0.0f;

    const char* AsBase = (const char*)&AsQ[0][0];
    const char* BsBase = (const char*)&BsQ[0][0];

    auto issueStage = [&](unsigned ioff, int kk) {
#pragma unroll
        for (int p = 0; p < 2; ++p) {
            ASYNC_CP_B128(ldsA + ioff + p * 4096,
                          (unsigned long long)(uintptr_t)(aptr[p] + kk));
            ASYNC_CP_B128(ldsB + ioff + p * 4096,
                          (unsigned long long)(uintptr_t)(bptr[p] + kk));
        }
    };

    auto computeStage = [&](unsigned coff) {
        const bf16_t* As = (const bf16_t*)(AsBase + coff);
        const bf16_t* Bs = (const bf16_t*)(BsBase + coff);
        FragB32 bfr[2];
#pragma unroll
        for (int j = 0; j < 2; ++j) {
            const char* bp = (const char*)(Bs + (wn * 32 + j * 16 + lr) * BK) + hl * 32;
            bfr[j].q[0] = *(const uint4*)bp;
            bfr[j].q[1] = *(const uint4*)(bp + 16);
        }
        FragB32 af[4];
#pragma unroll
        for (int i = 0; i < 4; ++i) {
            const char* ap = (const char*)(As + (wm * 64 + i * 16 + lr) * BK) + hl * 32;
            af[i].q[0] = *(const uint4*)ap;
            af[i].q[1] = *(const uint4*)(ap + 16);
        }
#pragma unroll
        for (int i = 0; i < 4; ++i) {
            acc[i][0] = __builtin_amdgcn_wmma_f32_16x16x32_bf16(
                false, af[i].v, false, bfr[0].v, (short)0, acc[i][0], false, false);
            acc[i][1] = __builtin_amdgcn_wmma_f32_16x16x32_bf16(
                false, af[i].v, false, bfr[1].v, (short)0, acc[i][1], false, false);
        }
    };

    const int nstages = K / BK;          // 160 (up) / 640 (down), always >= 2

    // prologue: issue stages 0 and 1
    issueStage(0, 0);
    issueStage(BUFB, BK);

    unsigned cOff = 0;            // compute-stage buffer byte offset
    unsigned iOff = 2 * BUFB;     // issue-stage   buffer byte offset

    // steady state: stages [0, nstages-2), branch-free body
    for (int s = 0; s < nstages - 2; ++s) {
        WAIT_ASYNC(0x4);          // stage s landed; stage s+1 still in flight
        __syncthreads();          // visible to all waves; s-1 reads complete
        issueStage(iOff, (s + 2) * BK);
        iOff = (iOff == (NBUF - 1) * BUFB) ? 0u : iOff + BUFB;
        computeStage(cOff);
        cOff = (cOff == (NBUF - 1) * BUFB) ? 0u : cOff + BUFB;
    }
    // drain stage nstages-2 (stage nstages-1 still in flight)
    WAIT_ASYNC(0x4);
    __syncthreads();
    computeStage(cOff);
    cOff = (cOff == (NBUF - 1) * BUFB) ? 0u : cOff + BUFB;
    // drain stage nstages-1 (last)
    WAIT_ASYNC(0x0);
    __syncthreads();
    computeStage(cOff);

    // epilogue: C layout = VGPR r -> (M = r + 8*half, N = lr)
    bf16_t* outcol = Out + (size_t)(n0 + wn * 32 + lr);
#pragma unroll
    for (int i = 0; i < 4; ++i) {
#pragma unroll
        for (int r = 0; r < 8; ++r) {
            const int m = m0 + wm * 64 + i * 16 + hl * 8 + r;
            if (m < cnt) {
                const size_t orow = GELU ? (size_t)(base + m) : (size_t)perm[base + m];
                bf16_t* op = outcol + orow * (size_t)ONum;
#pragma unroll
                for (int j = 0; j < 2; ++j) {
                    float v = acc[i][j][r];
                    if (GELU) {
                        float g = fminf(v, 7.0f);
                        v = g * __builtin_amdgcn_rcpf(1.0f + __expf(-1.702f * g));
                    }
                    op[j * 16] = (bf16_t)v;
                }
            }
        }
    }
}

// ---------------------------------------------------------------- launch
extern "C" void kernel_launch(void* const* d_in, const int* in_sizes, int n_in,
                              void* d_out, int out_size, void* d_ws, size_t ws_size,
                              hipStream_t stream)
{
    (void)in_sizes; (void)n_in; (void)out_size; (void)ws_size;

    const bf16_t* x      = (const bf16_t*)d_in[0];
    const int*    mod    = (const int*)d_in[1];
    const bf16_t* norm_w = (const bf16_t*)d_in[2];
    const bf16_t* up_w   = (const bf16_t*)d_in[3];
    const bf16_t* down_w = (const bf16_t*)d_in[4];
    bf16_t*       out    = (bf16_t*)d_out;

    char* ws = (char*)d_ws;
    int* counts = (int*)(ws + 0);
    int* offs   = (int*)(ws + 64);
    int* cursor = (int*)(ws + 128);
    int* perm   = (int*)(ws + 256);
    bf16_t* xn  = (bf16_t*)(ws + 65536);                                   // [N,H] bf16
    bf16_t* hg  = (bf16_t*)(ws + 65536 + (size_t)N_TOK * HID * 2);         // [N,I] bf16

    init_counts<<<1, 32, 0, stream>>>(counts);
    count_kernel<<<N_TOK / 256, 256, 0, stream>>>(mod, counts);
    scan_kernel<<<1, 32, 0, stream>>>(counts, offs, cursor);
    rmsnorm_gather<<<N_TOK, 256, 0, stream>>>(x, mod, norm_w, xn, cursor, perm);

    dim3 gu(INTER / BN, N_TOK / BM, NE);
    moe_gemm<true><<<gu, THREADS, 0, stream>>>(xn, up_w, hg, offs, counts, perm,
                                               HID, INTER, INTER);
    dim3 gd(HID / BN, N_TOK / BM, NE);
    moe_gemm<false><<<gd, THREADS, 0, stream>>>(hg, down_w, out, offs, counts, perm,
                                                INTER, HID, HID);
}
